// Cosine_Mat_15015205667291
// MI455X (gfx1250) — matrix operand order repaired and checked by reading, not compiled, not run on hardware
//
#include <hip/hip_runtime.h>

// ---------------------------------------------------------------------------
// Pairwise cosine distance:  out[i][j] = 1 - (a_i . a_j) / max(|a_i||a_j|, eps)
// mapping: [8192, 512] fp32, out: [8192, 8192] fp32.
//
// Strategy (MI455X / gfx1250, wave32, WMMA):
//   1) prep kernel: row norms (fp32) + bf16 hi/lo split of A into workspace.
//   2) GEMM A.A^T via bf16x3 split on V_WMMA_F32_16X16X32_BF16 (fp32-class
//      accuracy, ~2.7x fewer matrix-op issues than the f32 K=4 WMMA path).
//   3) Epilogue divides by the norm outer product and writes coalesced rows.
//   Fallback (tiny workspace): exact V_WMMA_F32_16X16X4_F32 GEMM on fp32 A.
// ---------------------------------------------------------------------------

#define N_DIM 8192
#define K_DIM 512
#define EPS   1e-8f

typedef __attribute__((ext_vector_type(16))) __bf16        v16bf;
typedef __attribute__((ext_vector_type(8)))  float         v8f;
typedef __attribute__((ext_vector_type(2)))  float         v2f;
typedef __attribute__((ext_vector_type(4)))  unsigned int  v4u;   // 16 bytes

union FragBF { v16bf v; v4u u[2]; };

// ---- bf16 conversion via bit ops (round-to-nearest-even), no __bf16 arith --
__device__ __forceinline__ unsigned short f32_to_bf16_rne(float x) {
  unsigned u = __float_as_uint(x);
  u += 0x7FFFu + ((u >> 16) & 1u);
  return (unsigned short)(u >> 16);
}
__device__ __forceinline__ float bf16_bits_to_f32(unsigned short s) {
  return __uint_as_float(((unsigned)s) << 16);
}

// ---------------------------------------------------------------------------
// Kernel 1: row sum-of-squares -> norms, plus optional bf16 hi/lo split.
// One wave per row (8 rows / 256-thread block). Lane-strided = coalesced.
// ---------------------------------------------------------------------------
__global__ __launch_bounds__(256)
void cosmat_prep(const float* __restrict__ A,
                 float* __restrict__ norms,
                 unsigned short* __restrict__ Ahi,
                 unsigned short* __restrict__ Alo,
                 int do_split) {
  const int row  = blockIdx.x * 8 + (threadIdx.x >> 5);
  const int lane = threadIdx.x & 31;
  const float* r = A + (size_t)row * K_DIM;

  float ss = 0.0f;
  #pragma unroll
  for (int c = lane; c < K_DIM; c += 32) {
    float x = r[c];
    ss = fmaf(x, x, ss);
    if (do_split) {
      unsigned short h = f32_to_bf16_rne(x);
      float          hf = bf16_bits_to_f32(h);
      unsigned short l = f32_to_bf16_rne(x - hf);
      Ahi[(size_t)row * K_DIM + c] = h;
      Alo[(size_t)row * K_DIM + c] = l;
    }
  }
  // wave32 butterfly reduction
  #pragma unroll
  for (int off = 16; off > 0; off >>= 1) ss += __shfl_xor(ss, off, 32);
  if (lane == 0) norms[row] = sqrtf(ss);
}

// ---------------------------------------------------------------------------
// Fragment loads. Because B = A^T of the column block, both A- and B-fragments
// are gathered from *rows* of A, with the same per-lane K chunking
// (ISA 7.12.2 layouts):
//   A 16x32 bf16: lane (h=lane>>4) holds K = {k0+8h .. +7} and {k0+16+8h .. +7}
//   B 32x16 bf16: lane holds K = {k0+8h .. +7} and {k0+16+8h .. +7} (same as A)
// ---------------------------------------------------------------------------
__device__ __forceinline__ v16bf load_frag_a_bf(const unsigned short* rowp,
                                                int k0, int h) {
  FragBF f;
  const unsigned short* p = rowp + k0 + h * 8;
  f.u[0] = *(const v4u*)(p);
  f.u[1] = *(const v4u*)(p + 16);
  return f.v;
}
__device__ __forceinline__ v16bf load_frag_b_bf(const unsigned short* rowp,
                                                int k0, int h) {
  FragBF f;
  const unsigned short* p = rowp + k0 + h * 8;
  f.u[0] = *(const v4u*)(p);
  f.u[1] = *(const v4u*)(p + 16);
  return f.v;
}

// ---------------------------------------------------------------------------
// Epilogue for one 16x16 f32 accumulator tile.
// C layout: VGPR r, lane = h*16 + c  ->  C[M = r + 8h][N = c].
// Row-segment stores: lanes 0-15 and 16-31 each write 64 contiguous bytes.
// ---------------------------------------------------------------------------
__device__ __forceinline__ void cosmat_store_tile(float* __restrict__ out,
                                                  const float* __restrict__ norms,
                                                  v8f c, int Mt, int Nt, int lane) {
  const int h = lane >> 4, l16 = lane & 15;
  const float nj = norms[Nt + l16];
  const float* np = norms + Mt + h * 8;      // 8 row norms for this lane-half
  #pragma unroll
  for (int r = 0; r < 8; ++r) {
    float ni    = np[r];
    float denom = fmaxf(ni * nj, EPS);
    float val   = 1.0f - c[r] / denom;
    out[(size_t)(Mt + h * 8 + r) * N_DIM + Nt + l16] = val;
  }
}

// ---------------------------------------------------------------------------
// Kernel 2a: bf16x3 split GEMM.  Workgroup = 8 waves covering 128(M) x 64(N);
// each wave owns a 32x32 patch = 2x2 tiles of 16x16. K loop steps by 32;
// 12 v_wmma_f32_16x16x32_bf16 per step per wave.
// ---------------------------------------------------------------------------
__global__ __launch_bounds__(256)
void cosmat_gemm_bf16x3(const unsigned short* __restrict__ Ahi,
                        const unsigned short* __restrict__ Alo,
                        const float* __restrict__ norms,
                        float* __restrict__ out) {
  const int lane = threadIdx.x & 31;
  const int wave = threadIdx.x >> 5;
  const int wr = wave & 3;              // 4 waves along M
  const int wc = wave >> 2;             // 2 waves along N
  const int Mbase = blockIdx.y * 128 + wr * 32;
  const int Nbase = blockIdx.x * 64  + wc * 32;
  const int h   = lane >> 4;
  const int l16 = lane & 15;

  size_t rowA[2], rowB[2];
  #pragma unroll
  for (int m = 0; m < 2; ++m) rowA[m] = (size_t)(Mbase + m * 16 + l16) * K_DIM;
  #pragma unroll
  for (int n = 0; n < 2; ++n) rowB[n] = (size_t)(Nbase + n * 16 + l16) * K_DIM;

  v8f acc[2][2];
  #pragma unroll
  for (int m = 0; m < 2; ++m)
    #pragma unroll
    for (int n = 0; n < 2; ++n) acc[m][n] = (v8f)(0.0f);

  for (int k0 = 0; k0 < K_DIM; k0 += 32) {
    // prefetch next K-chunk of this wave's A rows into near caches
    if (k0 + 32 < K_DIM) {
      __builtin_prefetch(Ahi + rowA[0] + k0 + 32, 0, 1);
      __builtin_prefetch(Ahi + rowB[0] + k0 + 32, 0, 1);
    }
    v16bf ah[2], al[2], bh[2], bl[2];
    #pragma unroll
    for (int m = 0; m < 2; ++m) {
      ah[m] = load_frag_a_bf(Ahi + rowA[m], k0, h);
      al[m] = load_frag_a_bf(Alo + rowA[m], k0, h);
    }
    #pragma unroll
    for (int n = 0; n < 2; ++n) {
      bh[n] = load_frag_b_bf(Ahi + rowB[n], k0, h);
      bl[n] = load_frag_b_bf(Alo + rowB[n], k0, h);
    }
    #pragma unroll
    for (int m = 0; m < 2; ++m)
      #pragma unroll
      for (int n = 0; n < 2; ++n) {
        // small cross terms first, dominant term last
        acc[m][n] = __builtin_amdgcn_wmma_f32_16x16x32_bf16(
            false, al[m], false, bh[n], (short)0, acc[m][n], false, false);
        acc[m][n] = __builtin_amdgcn_wmma_f32_16x16x32_bf16(
            false, ah[m], false, bl[n], (short)0, acc[m][n], false, false);
        acc[m][n] = __builtin_amdgcn_wmma_f32_16x16x32_bf16(
            false, ah[m], false, bh[n], (short)0, acc[m][n], false, false);
      }
  }

  #pragma unroll
  for (int m = 0; m < 2; ++m)
    #pragma unroll
    for (int n = 0; n < 2; ++n)
      cosmat_store_tile(out, norms, acc[m][n], Mbase + m * 16, Nbase + n * 16, lane);
}

// ---------------------------------------------------------------------------
// Kernel 2b: exact fp32 fallback GEMM via V_WMMA_F32_16X16X4_F32.
// A 16x4 f32 fragment: lane holds {A[row][k0+2h], A[row][k0+2h+1]}; B is the
// mirror layout from column-block rows (identical per-lane addressing).
// ---------------------------------------------------------------------------
__global__ __launch_bounds__(256)
void cosmat_gemm_f32(const float* __restrict__ A,
                     const float* __restrict__ norms,
                     float* __restrict__ out) {
  const int lane = threadIdx.x & 31;
  const int wave = threadIdx.x >> 5;
  const int wr = wave & 3;
  const int wc = wave >> 2;
  const int Mbase = blockIdx.y * 128 + wr * 32;
  const int Nbase = blockIdx.x * 64  + wc * 32;
  const int h   = lane >> 4;
  const int l16 = lane & 15;

  const float* rowA[2];
  const float* rowB[2];
  #pragma unroll
  for (int m = 0; m < 2; ++m) rowA[m] = A + (size_t)(Mbase + m * 16 + l16) * K_DIM;
  #pragma unroll
  for (int n = 0; n < 2; ++n) rowB[n] = A + (size_t)(Nbase + n * 16 + l16) * K_DIM;

  v8f acc[2][2];
  #pragma unroll
  for (int m = 0; m < 2; ++m)
    #pragma unroll
    for (int n = 0; n < 2; ++n) acc[m][n] = (v8f)(0.0f);

  for (int k0 = 0; k0 < K_DIM; k0 += 4) {
    v2f a[2], b[2];
    #pragma unroll
    for (int m = 0; m < 2; ++m) a[m] = *(const v2f*)(rowA[m] + k0 + h * 2);
    #pragma unroll
    for (int n = 0; n < 2; ++n) b[n] = *(const v2f*)(rowB[n] + k0 + h * 2);
    #pragma unroll
    for (int m = 0; m < 2; ++m)
      #pragma unroll
      for (int n = 0; n < 2; ++n)
        acc[m][n] = __builtin_amdgcn_wmma_f32_16x16x4_f32(
            false, a[m], false, b[n], (short)0, acc[m][n], false, false);
  }

  #pragma unroll
  for (int m = 0; m < 2; ++m)
    #pragma unroll
    for (int n = 0; n < 2; ++n)
      cosmat_store_tile(out, norms, acc[m][n], Mbase + m * 16, Nbase + n * 16, lane);
}

// ---------------------------------------------------------------------------
extern "C" void kernel_launch(void* const* d_in, const int* in_sizes, int n_in,
                              void* d_out, int out_size, void* d_ws, size_t ws_size,
                              hipStream_t stream) {
  const float* mapping = (const float*)d_in[0];
  float* out = (float*)d_out;

  // workspace layout: [norms: 32KB][Ahi: 8MB][Alo: 8MB]
  const size_t normBytes  = (size_t)N_DIM * sizeof(float);          // 32 KB
  const size_t splitElems = (size_t)N_DIM * K_DIM;
  const size_t splitBytes = splitElems * sizeof(unsigned short);    // 8 MB each
  char* ws = (char*)d_ws;
  float* norms = (float*)ws;
  unsigned short* Ahi = (unsigned short*)(ws + ((normBytes + 255) & ~(size_t)255));
  unsigned short* Alo = Ahi + splitElems;
  const int do_split =
      (ws_size >= ((normBytes + 255) & ~(size_t)255) + 2 * splitBytes) ? 1 : 0;

  cosmat_prep<<<N_DIM / 8, 256, 0, stream>>>(mapping, norms, Ahi, Alo, do_split);

  dim3 grid(N_DIM / 64, N_DIM / 128);   // 128 x 64 workgroups
  if (do_split) {
    cosmat_gemm_bf16x3<<<grid, 256, 0, stream>>>(Ahi, Alo, norms, out);
  } else {
    cosmat_gemm_f32<<<grid, 256, 0, stream>>>(mapping, norms, out);
  }
}
